// CBSA_9517647528038
// MI455X (gfx1250) — matrix operand order repaired and checked by reading, not compiled, not run on hardware
//
#include <hip/hip_runtime.h>
#include <hip/hip_bf16.h>

// ---------------------------------------------------------------------------
// CBSA block for MI455X (gfx1250): all matmuls via v_wmma_f32_16x16x32_bf16.
// B operands pre-transposed in memory so every fragment is 2x 16B loads.
// ---------------------------------------------------------------------------

typedef __bf16 bf16;
typedef __attribute__((ext_vector_type(16))) __bf16 v16bf;
typedef __attribute__((ext_vector_type(8)))  __bf16 bf16x8;
typedef __attribute__((ext_vector_type(8)))  float  v8f;

static constexpr int   kB     = 64;
static constexpr int   kH     = 12;
static constexpr int   kN     = 785;   // 1 cls + 784 patches
static constexpr int   kNP    = 784;
static constexpr int   kD     = 768;   // HEADS*DH
static constexpr int   kDH    = 64;
static constexpr int   kR     = 64;    // REP*REP
static constexpr float kSCALE = 0.125f;

__device__ __forceinline__ v8f wmma_bf16(v16bf a, v16bf b, v8f c) {
  return __builtin_amdgcn_wmma_f32_16x16x32_bf16(
      false, a, false, b, (short)0, c, false, false);
}

__device__ __forceinline__ bf16x8 bz8() {
  bf16x8 z;
#pragma unroll
  for (int j = 0; j < 8; ++j) z[j] = (bf16)0.f;
  return z;
}

// A fragment, operand row-major (K contiguous): v16bf = one contiguous
// 16-element run at K = 16*half.
__device__ __forceinline__ v16bf frag_a_rm(const bf16* p, int ld, int lane) {
  const int half = (lane >> 4) & 1, m = lane & 15;
  const bf16* q = p + m * ld + 16 * half;
  bf16x8 lo = *(const bf16x8*)q;
  bf16x8 hi = *(const bf16x8*)(q + 8);
  v16bf r;
#pragma unroll
  for (int j = 0; j < 8; ++j) { r[j] = lo[j]; r[8 + j] = hi[j]; }
  return r;
}
__device__ __forceinline__ v16bf frag_a_rm_tail(const bf16* p, int ld, int lane, bool full) {
  const int half = (lane >> 4) & 1, m = lane & 15;
  const bf16* q = p + m * ld + 16 * half;
  const bool ok = full || half == 0;
  bf16x8 lo = ok ? *(const bf16x8*)q : bz8();
  bf16x8 hi = ok ? *(const bf16x8*)(q + 8) : bz8();
  v16bf r;
#pragma unroll
  for (int j = 0; j < 8; ++j) { r[j] = lo[j]; r[8 + j] = hi[j]; }
  return r;
}
// B fragment, operand column-major: element (k,n) at p[n*ld + k];
// per lane one contiguous 16-element run at K = 16*half.
__device__ __forceinline__ v16bf frag_b_cm(const bf16* p, int ld, int lane) {
  const int half = (lane >> 4) & 1, n = lane & 15;
  const bf16* q = p + n * ld + 16 * half;
  bf16x8 lo = *(const bf16x8*)q;
  bf16x8 hi = *(const bf16x8*)(q + 8);
  v16bf r;
#pragma unroll
  for (int j = 0; j < 8; ++j) { r[j] = lo[j]; r[8 + j] = hi[j]; }
  return r;
}
// Tail variant: when only K < 16 remain, half==1 lanes' run is fully OOB.
__device__ __forceinline__ v16bf frag_b_cm_tail(const bf16* p, int ld, int lane, bool full) {
  const int half = (lane >> 4) & 1, n = lane & 15;
  v16bf r;
  if (full || half == 0) {
    const bf16* q = p + n * ld + 16 * half;
    bf16x8 lo = *(const bf16x8*)q;
    bf16x8 hi = *(const bf16x8*)(q + 8);
#pragma unroll
    for (int j = 0; j < 8; ++j) { r[j] = lo[j]; r[8 + j] = hi[j]; }
  } else {
#pragma unroll
    for (int j = 0; j < 16; ++j) r[j] = (bf16)0.f;
  }
  return r;
}
// C/D 16x16 f32: element v <-> row v + 8*half, col = lane & 15.

// ---------------------------------------------------------------------------
// Kernel 0: W[k][n] f32  ->  Wt[n][k] bf16  (run once per weight; L2-resident)
// ---------------------------------------------------------------------------
__global__ __launch_bounds__(256) void k_transpose_w(
    const float* __restrict__ W, bf16* __restrict__ Wt) {
  __shared__ bf16 sT[32][33];
  const int kBase = blockIdx.y * 32, nBase = blockIdx.x * 32;
  const int rr = threadIdx.x >> 5, cc = threadIdx.x & 31;
#pragma unroll
  for (int i = 0; i < 4; ++i) {
    const int row = rr + i * 8;
    sT[cc][row] = (bf16)W[(size_t)(kBase + row) * kD + nBase + cc];
  }
  __syncthreads();
#pragma unroll
  for (int i = 0; i < 4; ++i) {
    const int row = rr + i * 8;
    Wt[(size_t)(nBase + row) * kD + kBase + cc] = sT[row][cc];
  }
}

// ---------------------------------------------------------------------------
// Kernel 1: inner = x @ W_in  ->  inner[b,h,785,64] bf16 (+ patchesT copy).
// Block 64(M)x256(N) (3 N-blocks -> x re-read only 3x), 8 waves in 2x4,
// each wave a 2x4 grid of WMMA tiles (8 wmma / K-step).
// ---------------------------------------------------------------------------
__global__ __launch_bounds__(256) void k_proj_in(
    const float* __restrict__ x, const bf16* __restrict__ WinT,
    bf16* __restrict__ inner, bf16* __restrict__ patchesT) {
  __shared__ __align__(16) bf16 sA[64 * 32];
  const int tid = threadIdx.x, lane = tid & 31, wave = tid >> 5;
  const int wm = wave >> 2, wn = wave & 3;
  const int mBase = blockIdx.y * 64, nBase = blockIdx.x * 256;
  const int bBase = mBase / kN, tBase = mBase % kN;
  const v8f vzero = {0.f, 0.f, 0.f, 0.f, 0.f, 0.f, 0.f, 0.f};
  v8f acc[2][4];
#pragma unroll
  for (int i = 0; i < 2; ++i)
#pragma unroll
    for (int j = 0; j < 4; ++j) acc[i][j] = vzero;

  for (int kk = 0; kk < kD; kk += 32) {
    {  // stage A tile 64x32, f32 -> bf16
      const int i = tid >> 2, jb = (tid & 3) * 8;
      const float4* src = (const float4*)&x[(size_t)(mBase + i) * kD + kk + jb];
      const float4 f0 = src[0], f1 = src[1];
      bf16x8 v;
      v[0] = (bf16)f0.x; v[1] = (bf16)f0.y; v[2] = (bf16)f0.z; v[3] = (bf16)f0.w;
      v[4] = (bf16)f1.x; v[5] = (bf16)f1.y; v[6] = (bf16)f1.z; v[7] = (bf16)f1.w;
      *(bf16x8*)&sA[i * 32 + jb] = v;
      if (kk + 32 < kD) __builtin_prefetch((const char*)src + 128, 0, 1);
    }
    __syncthreads();
    v16bf a0 = frag_a_rm(&sA[(wm * 32 + 0) * 32], 32, lane);
    v16bf a1 = frag_a_rm(&sA[(wm * 32 + 16) * 32], 32, lane);
#pragma unroll
    for (int j = 0; j < 4; ++j) {
      v16bf bj = frag_b_cm(WinT + (size_t)(nBase + wn * 64 + j * 16) * kD + kk, kD, lane);
      acc[0][j] = wmma_bf16(a0, bj, acc[0][j]);
      acc[1][j] = wmma_bf16(a1, bj, acc[1][j]);
    }
    __syncthreads();
  }
  const int half = (lane >> 4) & 1, col = lane & 15;
#pragma unroll
  for (int i = 0; i < 2; ++i)
#pragma unroll
    for (int j = 0; j < 4; ++j) {
      const int c = nBase + wn * 64 + j * 16 + col, h = c >> 6, d = c & 63;
      const int tA = tBase + wm * 32 + i * 16 + 8 * half;
      int t0 = tA, b0_ = bBase;
      if (t0 >= kN) { t0 -= kN; ++b0_; }
      bf16x8 pk;
#pragma unroll
      for (int v = 0; v < 8; ++v) {
        int t = tA + v, b_ = bBase;
        if (t >= kN) { t -= kN; ++b_; }
        const bf16 val = (bf16)acc[i][j][v];
        inner[((size_t)(b_ * kH + h) * kN + t) * kDH + d] = val;
        pk[v] = val;
      }
      if (t0 >= 1 && t0 + 7 <= kNP) {  // contiguous patch run -> one b128 store
        *(bf16x8*)&patchesT[((size_t)(b0_ * kH + h) * kDH + d) * kNP + (t0 - 1)] = pk;
      } else {
#pragma unroll
        for (int v = 0; v < 8; ++v) {
          int t = tA + v, b_ = bBase;
          if (t >= kN) { t -= kN; ++b_; }
          if (t >= 1)
            patchesT[((size_t)(b_ * kH + h) * kDH + d) * kNP + (t - 1)] = pk[v];
        }
      }
    }
}

// ---------------------------------------------------------------------------
// Kernel 2: exact AdaptiveAvgPool1d^2 (28->8): reps0[b,h,64,64] f32.
// ---------------------------------------------------------------------------
__global__ __launch_bounds__(256) void k_pool(
    const bf16* __restrict__ inner, float* __restrict__ reps0) {
  const int bh = blockIdx.x;
  const size_t pOff = (size_t)bh * kN * kDH + kDH;  // skip cls row
  const size_t rOff = (size_t)bh * kR * kDH;
  for (int o = threadIdx.x; o < 512; o += 256) {
    const int rs = o >> 3, dg = (o & 7) * 8;
    const int r = rs >> 3, s = rs & 7;
    const int ps = r * 28 / 8, pe = ((r + 1) * 28 + 7) / 8;
    const int qs = s * 28 / 8, qe = ((s + 1) * 28 + 7) / 8;
    float ac[8];
#pragma unroll
    for (int j = 0; j < 8; ++j) ac[j] = 0.f;
    for (int p = ps; p < pe; ++p)
      for (int q = qs; q < qe; ++q) {
        const bf16x8 v = *(const bf16x8*)&inner[pOff + (size_t)(p * 28 + q) * kDH + dg];
#pragma unroll
        for (int j = 0; j < 8; ++j) ac[j] += (float)v[j];
      }
    const float inv = 1.f / (float)((pe - ps) * (qe - qs));
#pragma unroll
    for (int j = 0; j < 8; ++j) reps0[rOff + rs * kDH + dg + j] = ac[j] * inv;
  }
}

// ---------------------------------------------------------------------------
// Kernel 3: per-(b,h) attention; block = 256 threads (8 waves) per (b,h).
// Writes token deltas in place into inner rows 1..784.
// ---------------------------------------------------------------------------
__global__ __launch_bounds__(256) void k_attn(
    const float* __restrict__ reps0, bf16* __restrict__ inner,
    const bf16* __restrict__ patchesT,
    bf16* __restrict__ attnU, bf16* __restrict__ attnT,
    const float* __restrict__ step_x, const float* __restrict__ step_rep) {
  __shared__ __align__(16) bf16 sQ[kR * kDH];     // reps0 bf16 [r][d]
  __shared__ __align__(16) bf16 sR1[kR * kDH];    // reps1 [r][d]
  __shared__ __align__(16) bf16 sR1T[kR * kDH];   // reps1 [d][r]
  __shared__ __align__(16) bf16 sA2[kR * kR];     // exp attn2 scores [r][s]
  __shared__ __align__(16) bf16 sRtrT[kR * kDH];  // rep_to_rep scaled, [d][r]
  __shared__ float sSum[kR], sSum2[kR];
  const int tid = threadIdx.x, lane = tid & 31, wave = tid >> 5;
  const int bh = blockIdx.x, h = bh % kH;
  const size_t iOff = (size_t)bh * kN * kDH;       // inner slice
  const size_t pOff = iOff + kDH;                  // patches rows (t>=1)
  const size_t ptOff = (size_t)bh * kDH * kNP;     // patchesT slice
  const size_t rOff = (size_t)bh * kR * kDH;
  const size_t aOff = (size_t)bh * kR * kNP;
  const size_t tOff = (size_t)bh * kNP * kR;
  const float stepR = step_rep[h], stepX = step_x[h];
  const v8f vzero = {0.f, 0.f, 0.f, 0.f, 0.f, 0.f, 0.f, 0.f};
  const int half = (lane >> 4) & 1, col = lane & 15;

  for (int o = tid; o < kR * kDH; o += 256) sQ[o] = (bf16)reps0[rOff + o];
  if (tid < kR) { sSum[tid] = 0.f; sSum2[tid] = 0.f; }
  __syncthreads();

  // --- Phase A: e = exp(scale * reps0 @ patches^T); row sums accumulated in
  //     registers across strips, one batch of LDS atomics at the end.
  {
    float sumAcc[4][8];
#pragma unroll
    for (int mt = 0; mt < 4; ++mt)
#pragma unroll
      for (int v = 0; v < 8; ++v) sumAcc[mt][v] = 0.f;

    for (int strip = wave; strip < 49; strip += 8) {
      const int n0 = strip * 16;
      v8f acc[4];
#pragma unroll
      for (int i = 0; i < 4; ++i) acc[i] = vzero;
#pragma unroll
      for (int k0 = 0; k0 < kDH; k0 += 32) {
        v16bf bfr = frag_b_cm(inner + pOff + (size_t)n0 * kDH + k0, kDH, lane);
#pragma unroll
        for (int mt = 0; mt < 4; ++mt) {
          v16bf afr = frag_a_rm(sQ + mt * 16 * kDH + k0, kDH, lane);
          acc[mt] = wmma_bf16(afr, bfr, acc[mt]);
        }
      }
      const int n = n0 + col;
#pragma unroll
      for (int mt = 0; mt < 4; ++mt) {
        bf16x8 pk;
#pragma unroll
        for (int v = 0; v < 8; ++v) {
          const int r = mt * 16 + v + 8 * half;
          const float e = __expf(acc[mt][v] * kSCALE);
          sumAcc[mt][v] += e;
          const bf16 eb = (bf16)e;
          attnU[aOff + (size_t)r * kNP + n] = eb;
          pk[v] = eb;
        }
        *(bf16x8*)&attnT[tOff + (size_t)n * kR + mt * 16 + 8 * half] = pk;
      }
    }
#pragma unroll
    for (int mt = 0; mt < 4; ++mt)
#pragma unroll
      for (int v = 0; v < 8; ++v)
        atomicAdd(&sSum[mt * 16 + v + 8 * half], sumAcc[mt][v]);
  }
  __threadfence();
  __syncthreads();

  // --- Phase B: rep_delta = attnU @ patches; reps1 = reps0 + stepR*delta/sum.
  //     Both operands straight from global. Tile pair t=wave, wave+8 shares B.
  {
    const int mtA = wave >> 2, nt = wave & 3;
    const bf16* aP0 = attnU + aOff + (size_t)(mtA * 16) * kNP;
    const bf16* aP1 = attnU + aOff + (size_t)((mtA + 2) * 16) * kNP;
    const bf16* bP  = patchesT + ptOff + (size_t)(nt * 16) * kNP;
    v8f acc0 = vzero, acc1 = vzero;
    for (int k0 = 0; k0 < kNP; k0 += 32) {
      const bool full = (k0 + 32 <= kNP);
      v16bf a0 = frag_a_rm_tail(aP0 + k0, kNP, lane, full);
      v16bf a1 = frag_a_rm_tail(aP1 + k0, kNP, lane, full);
      v16bf bf_ = frag_b_cm_tail(bP + k0, kNP, lane, full);
      acc0 = wmma_bf16(a0, bf_, acc0);
      acc1 = wmma_bf16(a1, bf_, acc1);
    }
    const int d = nt * 16 + col;
    bf16x8 p0, p1;
#pragma unroll
    for (int v = 0; v < 8; ++v) {
      const int r0 = mtA * 16 + v + 8 * half;
      const float rep10 = reps0[rOff + (size_t)r0 * kDH + d] + stepR * acc0[v] / sSum[r0];
      const bf16 rb0 = (bf16)rep10;
      sR1[r0 * kDH + d] = rb0; p0[v] = rb0;
      const int r1 = (mtA + 2) * 16 + v + 8 * half;
      const float rep11 = reps0[rOff + (size_t)r1 * kDH + d] + stepR * acc1[v] / sSum[r1];
      const bf16 rb1 = (bf16)rep11;
      sR1[r1 * kDH + d] = rb1; p1[v] = rb1;
    }
    *(bf16x8*)&sR1T[d * kR + mtA * 16 + 8 * half] = p0;
    *(bf16x8*)&sR1T[d * kR + (mtA + 2) * 16 + 8 * half] = p1;
  }
  __syncthreads();

  // --- Phase C: attn2 = exp(scale * reps1 @ reps1^T); row sums.
  for (int t = wave; t < 16; t += 8) {
    const int mt = t >> 2, nt = t & 3;
    v8f acc = vzero;
#pragma unroll
    for (int k0 = 0; k0 < kDH; k0 += 32) {
      v16bf afr = frag_a_rm(sR1 + mt * 16 * kDH + k0, kDH, lane);
      v16bf bfr = frag_b_cm(sR1 + nt * 16 * kDH + k0, kDH, lane);
      acc = wmma_bf16(afr, bfr, acc);
    }
    const int s = nt * 16 + col;
#pragma unroll
    for (int v = 0; v < 8; ++v) {
      const int r = mt * 16 + v + 8 * half;
      const float e = __expf(acc[v] * kSCALE);
      atomicAdd(&sSum2[r], e);
      sA2[r * kR + s] = (bf16)e;
    }
  }
  __syncthreads();

  // --- Phase D: rtr = attn2U @ reps1, rows pre-scaled by 1/(sum2[r]*sum[r]).
  for (int t = wave; t < 16; t += 8) {
    const int mt = t >> 2, nt = t & 3;
    v8f acc = vzero;
#pragma unroll
    for (int k0 = 0; k0 < kR; k0 += 32) {
      v16bf afr = frag_a_rm(sA2 + mt * 16 * kR + k0, kR, lane);
      v16bf bfr = frag_b_cm(sR1T + nt * 16 * kR + k0, kR, lane);
      acc = wmma_bf16(afr, bfr, acc);
    }
    const int d = nt * 16 + col;
    bf16x8 pk;
#pragma unroll
    for (int v = 0; v < 8; ++v) {
      const int r = mt * 16 + v + 8 * half;
      pk[v] = (bf16)(acc[v] / (sSum2[r] * sSum[r]));
    }
    *(bf16x8*)&sRtrT[d * kR + mt * 16 + 8 * half] = pk;
  }
  __syncthreads();

  // --- Phase E: delta = attn^T @ rtr, written in place into inner rows 1+.
  for (int mtile = wave; mtile < 49; mtile += 8) {
    const int n0 = mtile * 16;
#pragma unroll
    for (int nt = 0; nt < 4; ++nt) {
      v8f acc = vzero;
#pragma unroll
      for (int k0 = 0; k0 < kR; k0 += 32) {
        v16bf afr = frag_a_rm(attnT + tOff + (size_t)n0 * kR + k0, kR, lane);
        v16bf bfr = frag_b_cm(sRtrT + nt * 16 * kR + k0, kR, lane);
        acc = wmma_bf16(afr, bfr, acc);
      }
      const int d = nt * 16 + col;
#pragma unroll
      for (int v = 0; v < 8; ++v) {
        const int n = n0 + v + 8 * half;
        inner[iOff + (size_t)(1 + n) * kDH + d] = (bf16)(stepX * acc[v]);
      }
    }
  }
}

// ---------------------------------------------------------------------------
// Kernel 4: out = x + tokens @ W_out  (tokens == inner after in-place delta).
// ---------------------------------------------------------------------------
__global__ __launch_bounds__(256) void k_proj_out(
    const float* __restrict__ x, const bf16* __restrict__ WoutT,
    const bf16* __restrict__ inner, float* __restrict__ out) {
  __shared__ __align__(16) bf16 sA[64 * 32];
  const int tid = threadIdx.x, lane = tid & 31, wave = tid >> 5;
  const int wm = wave >> 2, wn = wave & 3;
  const int mBase = blockIdx.y * 64, nBase = blockIdx.x * 256;
  const int bBase = mBase / kN, tBase = mBase % kN;
  const v8f vzero = {0.f, 0.f, 0.f, 0.f, 0.f, 0.f, 0.f, 0.f};
  v8f acc[2][4];
#pragma unroll
  for (int i = 0; i < 2; ++i)
#pragma unroll
    for (int j = 0; j < 4; ++j) acc[i][j] = vzero;

  for (int kk = 0; kk < kD; kk += 32) {
    {  // stage A tile: one b128 in, one b128 out
      const int i = tid >> 2, jb = (tid & 3) * 8;
      int t = tBase + i, b_ = bBase;
      if (t >= kN) { t -= kN; ++b_; }
      const int hh = (kk + jb) >> 6, dB = (kk + jb) & 63;
      *(bf16x8*)&sA[i * 32 + jb] =
          *(const bf16x8*)&inner[((size_t)(b_ * kH + hh) * kN + t) * kDH + dB];
    }
    __syncthreads();
    v16bf a0 = frag_a_rm(&sA[(wm * 32 + 0) * 32], 32, lane);
    v16bf a1 = frag_a_rm(&sA[(wm * 32 + 16) * 32], 32, lane);
#pragma unroll
    for (int j = 0; j < 4; ++j) {
      v16bf bj = frag_b_cm(WoutT + (size_t)(nBase + wn * 64 + j * 16) * kD + kk, kD, lane);
      acc[0][j] = wmma_bf16(a0, bj, acc[0][j]);
      acc[1][j] = wmma_bf16(a1, bj, acc[1][j]);
    }
    __syncthreads();
  }
  const int half = (lane >> 4) & 1, col = lane & 15;
#pragma unroll
  for (int i = 0; i < 2; ++i)
#pragma unroll
    for (int j = 0; j < 4; ++j) {
      const int c = nBase + wn * 64 + j * 16 + col;
#pragma unroll
      for (int v = 0; v < 8; ++v) {
        const size_t g = (size_t)(mBase + wm * 32 + i * 16 + v + 8 * half);
        out[g * kD + c] = x[g * kD + c] + acc[i][j][v];
      }
    }
}

// ---------------------------------------------------------------------------
extern "C" void kernel_launch(void* const* d_in, const int* in_sizes, int n_in,
                              void* d_out, int out_size, void* d_ws, size_t ws_size,
                              hipStream_t stream) {
  (void)in_sizes; (void)n_in; (void)out_size; (void)ws_size;
  const float* x        = (const float*)d_in[0];
  const float* Win      = (const float*)d_in[1];
  const float* Wout     = (const float*)d_in[2];
  const float* step_x   = (const float*)d_in[3];
  const float* step_rep = (const float*)d_in[4];

  char* ws = (char*)d_ws;
  size_t off = 0;
  auto alloc = [&](size_t bytes) -> void* {
    void* p = ws + off;
    off += (bytes + 255) & ~(size_t)255;
    return p;
  };
  bf16*  WinT     = (bf16*)alloc((size_t)kD * kD * sizeof(bf16));              // 1.2 MB
  bf16*  WoutT    = (bf16*)alloc((size_t)kD * kD * sizeof(bf16));              // 1.2 MB
  bf16*  inner    = (bf16*)alloc((size_t)kB * kH * kN * kDH * sizeof(bf16));   // 77 MB
  bf16*  patchesT = (bf16*)alloc((size_t)kB * kH * kDH * kNP * sizeof(bf16));  // 77 MB
  float* reps0    = (float*)alloc((size_t)kB * kH * kR * kDH * sizeof(float)); // 12.6 MB
  bf16*  attnU    = (bf16*)alloc((size_t)kB * kH * kR * kNP * sizeof(bf16));   // 77 MB
  bf16*  attnT    = (bf16*)alloc((size_t)kB * kH * kNP * kR * sizeof(bf16));   // 77 MB

  const dim3 blk(256);
  k_transpose_w<<<dim3(kD / 32, kD / 32), blk, 0, stream>>>(Win, WinT);
  k_transpose_w<<<dim3(kD / 32, kD / 32), blk, 0, stream>>>(Wout, WoutT);
  k_proj_in<<<dim3(kD / 256, (kB * kN) / 64), blk, 0, stream>>>(x, WinT, inner, patchesT);
  k_pool<<<dim3(kB * kH), blk, 0, stream>>>(inner, reps0);
  k_attn<<<dim3(kB * kH), blk, 0, stream>>>(reps0, inner, patchesT, attnU, attnT,
                                            step_x, step_rep);
  k_proj_out<<<dim3(kD / 256, (kB * kN) / 64), blk, 0, stream>>>(x, WoutT, inner,
                                                                 (float*)d_out);
}
